// GraphAttentionLayer_11897059410823
// MI455X (gfx1250) — matrix operand order repaired and checked by reading, not compiled, not run on hardware
//
#include <hip/hip_runtime.h>

typedef __attribute__((ext_vector_type(16))) _Float16 v16h;
typedef __attribute__((ext_vector_type(8)))  float    v8f;
typedef __attribute__((ext_vector_type(4)))  int      v4i;

#define N_NODES 8192
#define F_IN    128
#define U_OUT   64

// ---------------------------------------------------------------------------
// K1: t = X @ W + b  (f32, exact)   and   s2 = t @ a2  (f32)
// s1 and a_bias cancel inside the row-softmax, so they are never needed.
// ---------------------------------------------------------------------------
__global__ __launch_bounds__(256) void gat_linear(
    const float* __restrict__ X, const float* __restrict__ W,
    const float* __restrict__ b, const float* __restrict__ a_vec,
    float* __restrict__ t, float* __restrict__ s2)
{
    __shared__ float sW[F_IN * U_OUT];   // 32 KB
    __shared__ float sX[4 * F_IN];
    __shared__ float sRed[256];
    const int tid = threadIdx.x;
    const int r0  = blockIdx.x * 4;

    for (int i = tid; i < F_IN * U_OUT; i += 256) sW[i] = W[i];
    for (int i = tid; i < 4 * F_IN;     i += 256) sX[i] = X[(size_t)r0 * F_IN + i];
    __syncthreads();

    const int r = tid >> 6, c = tid & 63;
    float acc = 0.f;
#pragma unroll 8
    for (int k = 0; k < F_IN; ++k) acc = fmaf(sX[r * F_IN + k], sW[k * U_OUT + c], acc);
    acc += b[c];
    t[(size_t)(r0 + r) * U_OUT + c] = acc;

    sRed[tid] = acc * a_vec[U_OUT + c];   // a2 = a_vec[U:2U]
    __syncthreads();
    if (c == 0) {
        float s = 0.f;
        for (int k = 0; k < U_OUT; ++k) s += sRed[r * U_OUT + k];
        s2[r0 + r] = s;
    }
}

// ---------------------------------------------------------------------------
// K2: global max of s2 (single block) — safe softmax shift for every row.
// ---------------------------------------------------------------------------
__global__ __launch_bounds__(256) void gat_max(const float* __restrict__ s2,
                                               float* __restrict__ Mp)
{
    __shared__ float sm[256];
    const int tid = threadIdx.x;
    float m = -1e30f;
    for (int i = tid; i < N_NODES; i += 256) m = fmaxf(m, s2[i]);
    sm[tid] = m;
    __syncthreads();
    for (int s = 128; s > 0; s >>= 1) {
        if (tid < s) sm[tid] = fmaxf(sm[tid], sm[tid + s]);
        __syncthreads();
    }
    if (tid == 0) Mp[0] = sm[0];
}

// ---------------------------------------------------------------------------
// K3: w_j = exp(s2_j - M);  w_h = fp16(w);  t_wT[n][j] = fp16(w_j * t[j][n])
// (transposed so WMMA B-fragments are contiguous 32B per lane)
// ---------------------------------------------------------------------------
__global__ __launch_bounds__(256) void gat_scale(
    const float* __restrict__ t, const float* __restrict__ s2,
    const float* __restrict__ Mp,
    _Float16* __restrict__ w_h, _Float16* __restrict__ t_wT)
{
    __shared__ float    sw[64];
    __shared__ _Float16 tile[64][72];
    const int tid = threadIdx.x;
    const int jb  = blockIdx.x * 64;
    const float M = Mp[0];

    if (tid < 64) {
        float w = __expf(s2[jb + tid] - M);
        sw[tid] = w;
        w_h[jb + tid] = (_Float16)w;
    }
    __syncthreads();
    for (int idx = tid; idx < 64 * 64; idx += 256) {
        int jj = idx >> 6, n = idx & 63;                       // coalesced read of t
        tile[jj][n] = (_Float16)(t[(size_t)(jb + jj) * U_OUT + n] * sw[jj]);
    }
    __syncthreads();
    for (int idx = tid; idx < 64 * 64; idx += 256) {
        int n = idx >> 6, jj = idx & 63;                       // coalesced write of t_wT
        t_wT[(size_t)n * N_NODES + jb + jj] = tile[jj][n];
    }
}

// ---------------------------------------------------------------------------
// K4: out[i,:] = (sum_j adj_ij * w_j t[j,:]) / (sum_j adj_ij * w_j)
// One block = 16 rows; 8 waves split j (wave-interleaved 32-wide chunks).
// Per chunk: A = 0/1 mask fragment (fp16), 4 B fragments of w*t, plus a
// w-column B fragment so the denominator also runs on the tensor core.
// 5 x v_wmma_f32_16x16x32_f16 per chunk.  HBM-bound on the 256 MB adjacency.
// ---------------------------------------------------------------------------
__global__ __launch_bounds__(256) void gat_attend(
    const int* __restrict__ adj, const _Float16* __restrict__ t_wT,
    const _Float16* __restrict__ w_h, float* __restrict__ out)
{
    __shared__ float num[16 * 64];
    __shared__ float den[16];
    const int tid  = threadIdx.x;
    const int lane = tid & 31;
    const int wave = tid >> 5;
    const int i0   = blockIdx.x * 16;

    for (int i = tid; i < 16 * 64; i += 256) num[i] = 0.f;
    if (tid < 16) den[tid] = 0.f;
    __syncthreads();

    const int m  = lane & 15;   // A-fragment row (M) for this lane
    const int hi = lane >> 4;   // lane<16 holds K{0..15}; lane>=16 K{16..31}

    v8f c0 = {}, c1 = {}, c2 = {}, c3 = {}, cd = {};

    const int*      arowBase = adj + (size_t)(i0 + m) * N_NODES + hi * 16;
    const _Float16* bBase    = t_wT + (size_t)(lane & 15) * N_NODES;

    for (int ch = wave; ch < N_NODES / 32; ch += 8) {
        const int j0 = ch * 32;
        const int* ar = arowBase + j0;
        __builtin_prefetch(ar + 8 * 32, 0, 3);     // next chunk of the adj stream

        v4i q0 = *(const v4i*)(ar);
        v4i q1 = *(const v4i*)(ar + 4);
        v4i q2 = *(const v4i*)(ar + 8);
        v4i q3 = *(const v4i*)(ar + 12);
        int g[16] = { q0.x, q0.y, q0.z, q0.w, q1.x, q1.y, q1.z, q1.w,
                      q2.x, q2.y, q2.z, q2.w, q3.x, q3.y, q3.z, q3.w };
        v16h a;
#pragma unroll
        for (int u = 0; u < 16; ++u) a[u] = g[u] ? (_Float16)1.0f : (_Float16)0.0f;

        const _Float16* bp = bBase + j0 + hi * 16;          // 32B aligned, contiguous
        v16h b0 = *(const v16h*)(bp);
        v16h b1 = *(const v16h*)(bp + 16 * N_NODES);
        v16h b2 = *(const v16h*)(bp + 32 * N_NODES);
        v16h b3 = *(const v16h*)(bp + 48 * N_NODES);
        v16h bw = {};
        if ((lane & 15) == 0) bw = *(const v16h*)(w_h + j0 + hi * 16); // denom column

        c0 = __builtin_amdgcn_wmma_f32_16x16x32_f16(false, a, false, b0, (short)0, c0, false, false);
        c1 = __builtin_amdgcn_wmma_f32_16x16x32_f16(false, a, false, b1, (short)0, c1, false, false);
        c2 = __builtin_amdgcn_wmma_f32_16x16x32_f16(false, a, false, b2, (short)0, c2, false, false);
        c3 = __builtin_amdgcn_wmma_f32_16x16x32_f16(false, a, false, b3, (short)0, c3, false, false);
        cd = __builtin_amdgcn_wmma_f32_16x16x32_f16(false, a, false, bw, (short)0, cd, false, false);
    }

    // C/D layout: VGPR r, lane<16 -> M=r,N=lane ; lane>=16 -> M=r+8,N=lane-16
    const int Nn = lane & 15;
#pragma unroll
    for (int r = 0; r < 8; ++r) {
        const int mr = r + hi * 8;
        atomicAdd(&num[mr * 64 +  0 + Nn], c0[r]);
        atomicAdd(&num[mr * 64 + 16 + Nn], c1[r]);
        atomicAdd(&num[mr * 64 + 32 + Nn], c2[r]);
        atomicAdd(&num[mr * 64 + 48 + Nn], c3[r]);
        if (Nn == 0) atomicAdd(&den[mr], cd[r]);
    }
    __syncthreads();

    for (int idx = tid; idx < 16 * 64; idx += 256) {
        const int mr = idx >> 6;
        out[(size_t)(i0 + mr) * U_OUT + (idx & 63)] = num[idx] / den[mr];
    }
}

// ---------------------------------------------------------------------------
extern "C" void kernel_launch(void* const* d_in, const int* in_sizes, int n_in,
                              void* d_out, int out_size, void* d_ws, size_t ws_size,
                              hipStream_t stream)
{
    const float* X     = (const float*)d_in[0];   // node_features (8192,128)
    const int*   adj   = (const int*)  d_in[1];   // adj_matrix    (8192,8192)
    const float* W     = (const float*)d_in[2];   // W_t           (128,64)
    const float* b     = (const float*)d_in[3];   // b_t           (64,)
    const float* a_vec = (const float*)d_in[4];   // a_vec         (128,1)
    // d_in[5] = a_bias: cancels inside the row softmax -> unused.
    float* out = (float*)d_out;

    char* ws = (char*)d_ws;
    float*    t    = (float*)ws;    ws += (size_t)N_NODES * U_OUT * sizeof(float);
    float*    s2   = (float*)ws;    ws += (size_t)N_NODES * sizeof(float);
    float*    Mp   = (float*)ws;    ws += 256;
    _Float16* w_h  = (_Float16*)ws; ws += (size_t)N_NODES * sizeof(_Float16);
    _Float16* t_wT = (_Float16*)ws; /* U_OUT * N_NODES fp16 */

    gat_linear<<<N_NODES / 4,  256, 0, stream>>>(X, W, b, a_vec, t, s2);
    gat_max   <<<1,            256, 0, stream>>>(s2, Mp);
    gat_scale <<<N_NODES / 64, 256, 0, stream>>>(t, s2, Mp, w_h, t_wT);
    gat_attend<<<N_NODES / 16, 256, 0, stream>>>(adj, t_wT, w_h, out);
}
